// Model_67826123539057
// MI455X (gfx1250) — matrix operand order repaired and checked by reading, not compiled, not run on hardware
//
#include <hip/hip_runtime.h>
#include <hip/hip_bf16.h>

// ---------------------------------------------------------------------------
// Model dims (match reference)
// ---------------------------------------------------------------------------
#define BB     8
#define TT     2048
#define NTOK   (BB*TT)        // 16384 tokens
#define INDIM  128
#define DMODEL 160
#define FFN    320
#define HIDDEN 128
#define NHEADS 4
#define DH     40
#define WBAND  64

typedef _Float16 f16;
typedef __attribute__((ext_vector_type(16))) _Float16 v16h;
typedef __attribute__((ext_vector_type(8)))  _Float16 v8h;
typedef __attribute__((ext_vector_type(8)))  float    v8f;

// ---------------------------------------------------------------------------
// fp32 -> fp16 weight conversion
// ---------------------------------------------------------------------------
__global__ __launch_bounds__(256) void cvt_f32_f16(const float* __restrict__ s,
                                                   f16* __restrict__ d, int n) {
  int i = blockIdx.x * 256 + threadIdx.x;
  if (i < n) d[i] = (f16)s[i];
}

// ---------------------------------------------------------------------------
// Token-GEMM on WMMA:  out[m, n] = act( A[m,:] . W[n,:] + bias[n] (+res) )
// A: f16 [NTOK x kin], row stride lda (halfs). W: f16 [kout x kin] row-major.
// Each wave computes a 32(M) x 16*NT(N) macro-tile: 2 A fragments shared
// across NT n-tiles, NT B fragments shared across 2 m-tiles -> 2*NT
// independent V_WMMA_F32_16X16X32_F16 per k-step (32B/lane of loads per wmma at
// NT=2). Direct global loads land in the exact WMMA register layout; no LDS.
// ---------------------------------------------------------------------------
template <int NT>
__global__ __launch_bounds__(256)
void gemm_f16_wmma(const f16* __restrict__ A, int lda,
                   const f16* __restrict__ W, int kin, int kout,
                   const float* __restrict__ bias,
                   f16* __restrict__ outH, float* __restrict__ outF,
                   const float* __restrict__ residual,
                   int ldo, int colofs, int relu, int mtiles) {
  int wave  = blockIdx.x * (blockDim.x >> 5) + (threadIdx.x >> 5);
  int ntsup = (kout >> 4) / NT;                // n super-tiles
  int msup  = mtiles >> 1;                     // m super-tiles (MT=2)
  if (wave >= msup * ntsup) return;            // wave-uniform exit (EXEC all-ones)
  int tm = (wave / ntsup) * 2;
  int tn = (wave % ntsup) * NT;
  int lane = threadIdx.x & 31;
  int hi   = lane >> 4;                        // half-wave selector
  int lr   = lane & 15;

  const f16* arow0 = A + (size_t)(tm * 16 + lr) * lda;
  const f16* arow1 = arow0 + (size_t)16 * lda;
  const f16* wrow[NT];
#pragma unroll
  for (int ni = 0; ni < NT; ++ni)
    wrow[ni] = W + (size_t)((tn + ni) * 16 + lr) * kin;

  v8f acc[2][NT] = {};
  for (int k0 = 0; k0 < kin; k0 += 32) {
    // A fragment (16x32): elems 0..7 = K[k0+8*hi+i], 8..15 = K[k0+16+8*hi+i]
    v8h a00 = *(const v8h*)(arow0 + k0 + hi * 8);
    v8h a01 = *(const v8h*)(arow0 + k0 + 16 + hi * 8);
    v16h a0 = __builtin_shufflevector(a00, a01, 0,1,2,3,4,5,6,7,8,9,10,11,12,13,14,15);
    v8h a10 = *(const v8h*)(arow1 + k0 + hi * 8);
    v8h a11 = *(const v8h*)(arow1 + k0 + 16 + hi * 8);
    v16h a1 = __builtin_shufflevector(a10, a11, 0,1,2,3,4,5,6,7,8,9,10,11,12,13,14,15);
    // B fragment (32x16): elems 0..7 = W[n][k0+8*hi+i], 8..15 = W[n][k0+16+8*hi+i]
    v16h b[NT];
#pragma unroll
    for (int ni = 0; ni < NT; ++ni) {
      v8h b0 = *(const v8h*)(wrow[ni] + k0 + hi * 8);
      v8h b1 = *(const v8h*)(wrow[ni] + k0 + 16 + hi * 8);
      b[ni] = __builtin_shufflevector(b0, b1, 0,1,2,3,4,5,6,7,8,9,10,11,12,13,14,15);
    }
#pragma unroll
    for (int ni = 0; ni < NT; ++ni) {
      acc[0][ni] = __builtin_amdgcn_wmma_f32_16x16x32_f16(false, a0, false, b[ni],
                                                          (short)0, acc[0][ni], false, false);
      acc[1][ni] = __builtin_amdgcn_wmma_f32_16x16x32_f16(false, a1, false, b[ni],
                                                          (short)0, acc[1][ni], false, false);
    }
  }

  float bn[NT];
#pragma unroll
  for (int ni = 0; ni < NT; ++ni)
    bn[ni] = bias ? bias[(tn + ni) * 16 + lr] : 0.f;
#pragma unroll
  for (int mi = 0; mi < 2; ++mi) {
#pragma unroll
    for (int ni = 0; ni < NT; ++ni) {
      int col = (tn + ni) * 16 + lr + colofs;
#pragma unroll
      for (int j = 0; j < 8; ++j) {            // D layout: row = j + 8*hi, col = lr
        int row   = (tm + mi) * 16 + j + 8 * hi;
        float v   = acc[mi][ni][j] + bn[ni];
        if (relu) v = v > 0.f ? v : 0.f;
        size_t idx = (size_t)row * ldo + col;
        if (residual) v += residual[idx];
        if (outF) outF[idx] = v;
        if (outH) outH[idx] = (f16)v;
      }
    }
  }
}

// ---------------------------------------------------------------------------
// Dilated grouped conv (TCN branch conv1), 16 groups of 8 ch. out: f16 [NTOK,384]
// ---------------------------------------------------------------------------
__global__ __launch_bounds__(256)
void tcn_conv1(const float* __restrict__ x, const float* __restrict__ w,
               const float* __restrict__ bias, f16* __restrict__ out,
               int k, int dil, int pad, int colofs) {
  int id = blockIdx.x * 256 + threadIdx.x;     // NTOK*128 threads
  int o  = id & 127;
  int n  = id >> 7;
  int b  = n >> 11;
  int t  = n & (TT - 1);
  int g  = o >> 3;
  const float* xb = x + ((size_t)b * TT) * INDIM + g * 8;
  float acc = bias[o];
  for (int kk = 0; kk < k; ++kk) {
    int ti = t - pad + kk * dil;
    if (ti < 0 || ti >= TT) continue;
    const float* xp = xb + (size_t)ti * INDIM;
    const float* wp = w + (o * 8) * k + kk;    // w[o][icg][kk]
#pragma unroll
    for (int ic = 0; ic < 8; ++ic) acc += wp[ic * k] * xp[ic];
  }
  out[(size_t)n * 384 + colofs + o] = (f16)(acc > 0.f ? acc : 0.f);
}

// ---------------------------------------------------------------------------
// LayerNorm over 160 dims; one wave per token; optional sinusoid posenc add.
// ---------------------------------------------------------------------------
__global__ __launch_bounds__(256)
void ln160(const float* __restrict__ in, const float* __restrict__ g,
           const float* __restrict__ bta, float* __restrict__ outF,
           f16* __restrict__ outH, int addpe) {
  int row = blockIdx.x * 8 + (threadIdx.x >> 5);
  if (row >= NTOK) return;
  int lane = threadIdx.x & 31;
  const float* rp = in + (size_t)row * DMODEL;
  float v[5], s = 0.f;
#pragma unroll
  for (int i = 0; i < 5; ++i) { v[i] = rp[lane + i * 32]; s += v[i]; }
#pragma unroll
  for (int o = 16; o > 0; o >>= 1) s += __shfl_xor(s, o, 32);
  float mean = s * (1.f / 160.f);
  float var = 0.f;
#pragma unroll
  for (int i = 0; i < 5; ++i) { float d = v[i] - mean; var += d * d; }
#pragma unroll
  for (int o = 16; o > 0; o >>= 1) var += __shfl_xor(var, o, 32);
  float rstd = rsqrtf(var * (1.f / 160.f) + 1e-5f);
  int t = row & (TT - 1);
#pragma unroll
  for (int i = 0; i < 5; ++i) {
    int e = lane + i * 32;
    float y = (v[i] - mean) * rstd * g[e] + bta[e];
    if (addpe) {
      float div = __expf(-(float)(e & ~1) * (9.2103403719761836f / (float)DMODEL));
      float ang = (float)t * div;
      y += (e & 1) ? __cosf(ang) : __sinf(ang);
    }
    size_t idx = (size_t)row * DMODEL + e;
    if (outF) outF[idx] = y;
    if (outH) outH[idx] = (f16)y;
  }
}

// ---------------------------------------------------------------------------
// Banded attention (|t-s| <= 64), online softmax; one thread per (b, head, t).
// qkv: f16 [NTOK, 480] (q|k|v). out: f16 [NTOK, 160].
// ---------------------------------------------------------------------------
__global__ __launch_bounds__(128)
void attn_band(const f16* __restrict__ qkv, f16* __restrict__ out) {
  int blk    = blockIdx.x;
  int chunks = TT / 128;
  int bh     = blk / chunks;
  int chunk  = blk - bh * chunks;
  int b      = bh >> 2;
  int h      = bh & 3;
  int t      = chunk * 128 + threadIdx.x;
  size_t nrow = (size_t)(b * TT + t);
  const f16* qp = qkv + nrow * 480 + h * DH;
  float q[DH];
#pragma unroll
  for (int i = 0; i < DH; ++i) q[i] = (float)qp[i] * 0.158113883f;  // 1/sqrt(40)
  float m = -1e30f, l = 0.f, acc[DH];
#pragma unroll
  for (int i = 0; i < DH; ++i) acc[i] = 0.f;
  int s0 = t - WBAND; if (s0 < 0) s0 = 0;
  int s1 = t + WBAND; if (s1 > TT - 1) s1 = TT - 1;
  for (int s = s0; s <= s1; ++s) {
    const f16* kp = qkv + (size_t)(b * TT + s) * 480 + DMODEL + h * DH;
    const f16* vp = kp + DMODEL;
    float d = 0.f;
#pragma unroll
    for (int i = 0; i < DH; ++i) d += q[i] * (float)kp[i];
    float p;
    if (d > m) {
      float c = __expf(m - d);
      l *= c;
#pragma unroll
      for (int i = 0; i < DH; ++i) acc[i] *= c;
      m = d; p = 1.f;
    } else {
      p = __expf(d - m);
    }
    l += p;
#pragma unroll
    for (int i = 0; i < DH; ++i) acc[i] += p * (float)vp[i];
  }
  f16* op = out + nrow * DMODEL + h * DH;
  float inv = 1.f / l;
#pragma unroll
  for (int i = 0; i < DH; ++i) op[i] = (f16)(acc[i] * inv);
}

// ---------------------------------------------------------------------------
// Peak head conv1 (k=3 over channels D=160 -> 80) + ReLU, fp32 in/out.
// ---------------------------------------------------------------------------
__global__ __launch_bounds__(256)
void peak_conv1(const float* __restrict__ h, const float* __restrict__ w,
                const float* __restrict__ bias, float* __restrict__ out) {
  int id = blockIdx.x * 256 + threadIdx.x;     // NTOK*80 threads
  int c1 = id % 80;
  int n  = id / 80;
  int b  = n >> 11, t = n & (TT - 1);
  float acc = bias[c1];
  for (int kk = 0; kk < 3; ++kk) {
    int ti = t - 1 + kk;
    if (ti < 0 || ti >= TT) continue;
    const float* hp = h + (size_t)(b * TT + ti) * DMODEL;
    const float* wp = w + (size_t)c1 * DMODEL * 3 + kk;
    for (int c = 0; c < DMODEL; ++c) acc += wp[c * 3] * hp[c];
  }
  out[(size_t)n * 80 + c1] = acc > 0.f ? acc : 0.f;
}

// ---------------------------------------------------------------------------
// Final: y_main (head2) + 0.3*sigmoid(peak conv2), scaled. One thread/token.
// ---------------------------------------------------------------------------
__global__ __launch_bounds__(256)
void final_head(const f16* __restrict__ thead, const float* __restrict__ hw2,
                const float* __restrict__ hb2, const float* __restrict__ pk1,
                const float* __restrict__ pw2, const float* __restrict__ pb2,
                const float* __restrict__ scale, float* __restrict__ out) {
  int n = blockIdx.x * 256 + threadIdx.x;
  int b = n >> 11, t = n & (TT - 1);
  float ym = hb2[0];
  const f16* tp = thead + (size_t)n * 80;
  for (int c = 0; c < 80; ++c) ym += hw2[c] * (float)tp[c];
  float pkv = pb2[0];
  for (int kk = 0; kk < 3; ++kk) {
    int ti = t - 1 + kk;
    if (ti < 0 || ti >= TT) continue;
    const float* pp = pk1 + (size_t)(b * TT + ti) * 80;
    for (int c = 0; c < 80; ++c) pkv += pw2[c * 3 + kk] * pp[c];
  }
  float sg = 1.f / (1.f + __expf(-pkv));
  out[n] = (ym + 0.3f * sg) * scale[0];
}

// ---------------------------------------------------------------------------
// Host launch sequence
// ---------------------------------------------------------------------------
extern "C" void kernel_launch(void* const* d_in, const int* in_sizes, int n_in,
                              void* d_out, int out_size, void* d_ws, size_t ws_size,
                              hipStream_t stream) {
  (void)in_sizes; (void)n_in; (void)out_size; (void)ws_size;
  const float* P[66];
  for (int i = 0; i < 66; ++i) P[i] = (const float*)d_in[i];
  const float* x = P[0];

  char* ws = (char*)d_ws;
  size_t cur = 0;
  auto alloc = [&](size_t bytes) -> char* {
    char* r = ws + cur;
    cur += (bytes + 255) & ~(size_t)255;
    return r;
  };

  // activations
  f16*   buf1   = (f16*)  alloc((size_t)NTOK * 384 * 2);   // conv1 branch outs
  f16*   feats  = (f16*)  alloc((size_t)NTOK * 384 * 2);   // conv2 concat
  f16*   xm     = (f16*)  alloc((size_t)NTOK * 128 * 2);
  f16*   xf     = (f16*)  alloc((size_t)NTOK * 128 * 2);
  float* projo  = (float*)alloc((size_t)NTOK * DMODEL * 4);
  float* h      = (float*)alloc((size_t)NTOK * DMODEL * 4); // residual stream
  f16*   hn     = (f16*)  alloc((size_t)NTOK * DMODEL * 2);
  f16*   qkv    = (f16*)  alloc((size_t)NTOK * 480 * 2);
  f16*   attno  = (f16*)  alloc((size_t)NTOK * DMODEL * 2);
  f16*   ffh    = (f16*)  alloc((size_t)NTOK * FFN * 2);
  f16*   hh     = (f16*)  alloc((size_t)NTOK * DMODEL * 2);
  f16*   thead  = (f16*)  alloc((size_t)NTOK * 80 * 2);
  float* pk1    = (float*)alloc((size_t)NTOK * 80 * 4);
  // fp16 weights
  f16* wbr2[3]; for (int i = 0; i < 3; ++i) wbr2[i] = (f16*)alloc(128 * 128 * 2);
  f16* wmg1  = (f16*)alloc(128 * 384 * 2);
  f16* wmg2  = (f16*)alloc(128 * 128 * 2);
  f16* wproj = (f16*)alloc(DMODEL * 128 * 2);
  f16 *wqkv[3], *wout[3], *wff1[3], *wff2[3];
  for (int l = 0; l < 3; ++l) {
    wqkv[l] = (f16*)alloc(480 * DMODEL * 2);
    wout[l] = (f16*)alloc(DMODEL * DMODEL * 2);
    wff1[l] = (f16*)alloc(FFN * DMODEL * 2);
    wff2[l] = (f16*)alloc(DMODEL * FFN * 2);
  }
  f16* whead1 = (f16*)alloc(80 * DMODEL * 2);

  auto cvt = [&](const float* s, f16* d, int n) {
    cvt_f32_f16<<<(n + 255) / 256, 256, 0, stream>>>(s, d, n);
  };
  auto gemm = [&](const f16* A, int lda, const f16* W, int kin, int kout,
                  const float* bias, f16* oh, float* of, const float* res,
                  int ldo, int colofs, int relu) {
    int ntiles = kout / 16;
    if ((ntiles & 1) == 0) {
      int waves  = (NTOK / 32) * (ntiles / 2);
      int blocks = (waves + 7) / 8;
      gemm_f16_wmma<2><<<blocks, 256, 0, stream>>>(A, lda, W, kin, kout, bias, oh,
                                                   of, res, ldo, colofs, relu,
                                                   NTOK / 16);
    } else {
      int waves  = (NTOK / 32) * ntiles;
      int blocks = (waves + 7) / 8;
      gemm_f16_wmma<1><<<blocks, 256, 0, stream>>>(A, lda, W, kin, kout, bias, oh,
                                                   of, res, ldo, colofs, relu,
                                                   NTOK / 16);
    }
  };

  // ---- weight conversion ----
  for (int i = 0; i < 3; ++i) cvt(P[3 + 4 * i], wbr2[i], 128 * 128);
  cvt(P[13], wmg1, 128 * 384);
  cvt(P[15], wmg2, 128 * 128);
  cvt(P[17], wproj, DMODEL * 128);
  for (int l = 0; l < 3; ++l) {
    int L = 21 + 12 * l;
    cvt(P[L + 0], wqkv[l], 480 * DMODEL);
    cvt(P[L + 2], wout[l], DMODEL * DMODEL);
    cvt(P[L + 8], wff1[l], FFN * DMODEL);
    cvt(P[L + 10], wff2[l], DMODEL * FFN);
  }
  cvt(P[57], whead1, 80 * DMODEL);

  // ---- TCN front-end ----
  const int ks[3]   = {3, 5, 7};
  const int dils[3] = {1, 2, 4};
  for (int i = 0; i < 3; ++i) {
    int k = ks[i], d = dils[i];
    int pad = (k + (k - 1) * (d - 1)) / 2;
    tcn_conv1<<<(NTOK * 128) / 256, 256, 0, stream>>>(x, P[1 + 4 * i], P[2 + 4 * i],
                                                      buf1, k, d, pad, i * 128);
  }
  for (int i = 0; i < 3; ++i)  // conv2 1x1 (GEMM) into feats columns
    gemm(buf1 + i * 128, 384, wbr2[i], 128, 128, P[4 + 4 * i],
         feats, nullptr, nullptr, 384, i * 128, 1);
  gemm(feats, 384, wmg1, 384, 128, P[14], xm, nullptr, nullptr, 128, 0, 1);
  gemm(xm, 128, wmg2, 128, 128, P[16], xf, nullptr, nullptr, 128, 0, 0);

  // ---- proj + LN + posenc ----
  gemm(xf, 128, wproj, 128, DMODEL, P[18], nullptr, projo, nullptr, DMODEL, 0, 0);
  ln160<<<NTOK / 8, 256, 0, stream>>>(projo, P[19], P[20], h, nullptr, 1);

  // ---- transformer layers ----
  for (int l = 0; l < 3; ++l) {
    int L = 21 + 12 * l;
    ln160<<<NTOK / 8, 256, 0, stream>>>(h, P[L + 4], P[L + 5], nullptr, hn, 0);
    gemm(hn, DMODEL, wqkv[l], DMODEL, 480, P[L + 1], qkv, nullptr, nullptr, 480, 0, 0);
    attn_band<<<BB * NHEADS * (TT / 128), 128, 0, stream>>>(qkv, attno);
    gemm(attno, DMODEL, wout[l], DMODEL, DMODEL, P[L + 3],
         nullptr, h, h, DMODEL, 0, 0);                      // h += attn out-proj
    ln160<<<NTOK / 8, 256, 0, stream>>>(h, P[L + 6], P[L + 7], nullptr, hn, 0);
    gemm(hn, DMODEL, wff1[l], DMODEL, FFN, P[L + 9], ffh, nullptr, nullptr, FFN, 0, 1);
    gemm(ffh, FFN, wff2[l], FFN, DMODEL, P[L + 11],
         (l == 2) ? hh : nullptr, h, h, DMODEL, 0, 0);      // h += ffn
  }

  // ---- heads ----
  gemm(hh, DMODEL, whead1, DMODEL, 80, P[58], thead, nullptr, nullptr, 80, 0, 1);
  peak_conv1<<<(NTOK * 80) / 256, 256, 0, stream>>>(h, P[61], P[62], pk1);
  final_head<<<NTOK / 256, 256, 0, stream>>>(thead, P[59], P[60], pk1, P[63], P[64],
                                             P[65], (float*)d_out);
}
